// DeformableDeconv_54563264528405
// MI455X (gfx1250) — matrix operand order repaired and checked by reading, not compiled, not run on hardware
//
#include <hip/hip_runtime.h>
#include <math.h>

typedef __bf16 bf16_t;
typedef __attribute__((ext_vector_type(4)))  __bf16 v4bf;
typedef __attribute__((ext_vector_type(8)))  __bf16 v8bf;
typedef __attribute__((ext_vector_type(16))) __bf16 v16bf;
typedef __attribute__((ext_vector_type(8)))  float  v8f;
typedef __attribute__((ext_vector_type(4)))  float  v4f;

#define Bsz   4
#define Cch   256
#define Hh    96
#define Ww    96
#define Och   256
#define HW    (Hh * Ww)        // 9216
#define SPAT  (Bsz * HW)       // 36864
#define KDIM  (9 * Cch)        // 2304
#define KCHUNKS (KDIM / 32)    // 72
#define OFF_CH 18

// ---------------------------------------------------------------------------
// Kernel 1: 3x3 offset conv, NCHW f32 direct, LDS-tiled over input channels.
// Tile: 8 x 32 output positions per workgroup, all 18 output channels in regs.
// ---------------------------------------------------------------------------
__global__ void __launch_bounds__(256)
offset_conv_kernel(const float* __restrict__ x,
                   const float* __restrict__ w,      // (18,256,3,3)
                   const float* __restrict__ bias,   // (18,)
                   float* __restrict__ off)          // (4,18,96,96)
{
    const int tid = threadIdx.x;
    const int bi  = blockIdx.x;          // 4 * 12 * 3 = 144 blocks
    const int b   = bi / 36;
    const int t   = bi % 36;
    const int ho0 = (t / 3) * 8;
    const int wo0 = (t % 3) * 32;

    __shared__ float patch[10][34];
    __shared__ float wl[OFF_CH * 9];

    const int ph = tid >> 5;             // 0..7
    const int pw = tid & 31;             // 0..31

    float acc[OFF_CH];
#pragma unroll
    for (int o = 0; o < OFF_CH; ++o) acc[o] = 0.f;

    for (int c = 0; c < Cch; ++c) {
        for (int idx = tid; idx < 340; idx += 256) {
            int r   = idx / 34;
            int col = idx - r * 34;
            int gy  = ho0 - 1 + r;
            int gx  = wo0 - 1 + col;
            float v = 0.f;
            if (gy >= 0 && gy < Hh && gx >= 0 && gx < Ww)
                v = x[((size_t)(b * Cch + c) * Hh + gy) * Ww + gx];
            patch[r][col] = v;
        }
        if (tid < OFF_CH * 9) {
            int o   = tid / 9;
            int tap = tid - o * 9;
            wl[tid] = w[((size_t)o * Cch + c) * 9 + tap];
        }
        __syncthreads();

#pragma unroll
        for (int tap = 0; tap < 9; ++tap) {
            const int ky = tap / 3, kx = tap % 3;
            const float xv = patch[ph + ky][pw + kx];
#pragma unroll
            for (int o = 0; o < OFF_CH; ++o)
                acc[o] += xv * wl[o * 9 + tap];
        }
        __syncthreads();
    }

    const int ho = ho0 + ph, wo = wo0 + pw;
#pragma unroll
    for (int o = 0; o < OFF_CH; ++o)
        off[((size_t)(b * OFF_CH + o) * Hh + ho) * Ww + wo] = acc[o] + bias[o];
}

// ---------------------------------------------------------------------------
// Kernel 2: pack deform_w (O,C,3,3) f32 -> bf16, n-major K layout:
//   Bp2[n][tap*256 + c]  so each lane's WMMA B-fragment is 16 contiguous bf16.
// ---------------------------------------------------------------------------
__global__ void __launch_bounds__(256)
pack_weights_kernel(const float* __restrict__ w, bf16_t* __restrict__ Bp2)
{
    const int i = blockIdx.x * 256 + threadIdx.x;   // 256 * 2304 = 589824
    if (i >= Och * KDIM) return;
    const int n   = i / KDIM;
    const int k   = i - n * KDIM;
    const int tap = k >> 8;          // k / 256
    const int c   = k & 255;         // k % 256
    Bp2[i] = (bf16_t)w[((size_t)n * Cch + c) * 9 + tap];
}

// ---------------------------------------------------------------------------
// Kernel 3: fused bilinear-gather + implicit GEMM via WMMA bf16.
//   GEMM: M = 36864 spatial, N = 256 out-ch, K = 2304 (tap-major, c-minor).
//   Workgroup: 256 thr (8 wave32). Tile: 32 m x 256 n. Wave: 16 m x 4*(16 n).
//   Software pipeline: double-buffered A tile in LDS, ONE barrier per K-chunk;
//   gather for chunk k+1 issues before the WMMAs of chunk k.
// ---------------------------------------------------------------------------
__global__ void __launch_bounds__(256)
deform_wmma_kernel(const float*  __restrict__ x,     // (4,256,96,96)
                   const float*  __restrict__ off,   // (4,18,96,96)
                   const bf16_t* __restrict__ Bp2,   // [256][2304]
                   const float*  __restrict__ bias,  // (256,)
                   float* __restrict__ out)          // (4,256,96,96)
{
    const int tid  = threadIdx.x;
    const int lane = tid & 31;
    const int wave = tid >> 5;
    const int half = lane >> 4;      // 0/1 (WMMA lane half)
    const int l15  = lane & 15;

    const int mbase = blockIdx.x * 32;          // spatial tile base
    const int mw    = (wave & 1) * 16;          // wave m offset in tile
    const int nw0   = (wave >> 1) * 64;         // wave n base (4 subtiles of 16)

    __shared__ __attribute__((aligned(16))) bf16_t a_lds[2][32][40]; // dbl-buffered A
    __shared__ int   gidx[9][32][4];   // bilinear corner indices per (tap, m)
    __shared__ float gwgt[9][32][4];   // bilinear corner weights per (tap, m)
    __shared__ int   gplane[32];       // b*C*HW per m
    __shared__ __attribute__((aligned(16))) float out_sT[Och][36]; // [n][m]

    v8f acc[4];
#pragma unroll
    for (int j = 0; j < 4; ++j) acc[j] = (v8f){0.f,0.f,0.f,0.f,0.f,0.f,0.f,0.f};

    // A-staging role for this thread: m row + 4 consecutive channel columns
    const int am  = tid & 31;
    const int cl0 = (tid >> 5) << 2;

    // ---- precompute bilinear tables for ALL 9 taps (once) ----
    for (int idx = tid; idx < 9 * 32; idx += 256) {
        const int tap = idx >> 5;
        const int m   = idx & 31;
        const int s   = mbase + m;
        const int b   = s / HW;
        const int rem = s - b * HW;
        const int ho  = rem / Ww;
        const int wo  = rem - ho * Ww;
        const float dy = off[((size_t)(b * OFF_CH + tap * 2    ) * Hh + ho) * Ww + wo];
        const float dx = off[((size_t)(b * OFF_CH + tap * 2 + 1) * Hh + ho) * Ww + wo];
        const float py = (float)(ho - 1 + tap / 3) + dy;
        const float px = (float)(wo - 1 + tap % 3) + dx;
        const float y0f = floorf(py), x0f = floorf(px);
        const float ty = py - y0f, tx = px - x0f;
        const int y0 = (int)y0f, x0 = (int)x0f;
        const float ws[4] = { (1.f - ty) * (1.f - tx), (1.f - ty) * tx,
                              ty * (1.f - tx),         ty * tx };
#pragma unroll
        for (int j = 0; j < 4; ++j) {
            const int yc = y0 + (j >> 1);
            const int xc = x0 + (j & 1);
            const bool valid = (yc >= 0) & (yc < Hh) & (xc >= 0) & (xc < Ww);
            const int yi = min(max(yc, 0), Hh - 1);
            const int xi = min(max(xc, 0), Ww - 1);
            gidx[tap][m][j] = yi * Ww + xi;
            gwgt[tap][m][j] = valid ? ws[j] : 0.f;
        }
        if (tap == 0) gplane[m] = b * Cch * HW;
    }
    __syncthreads();

    // stage one 32m x 32k A chunk (k = tap*8 + cblock) into buffer `buf`
    auto stageA = [&](int buf, int k) {
        const int tap = k >> 3;
        const int c0  = (k & 7) << 5;
        const int   xb = gplane[am];
        const int   i0 = gidx[tap][am][0], i1 = gidx[tap][am][1];
        const int   i2 = gidx[tap][am][2], i3 = gidx[tap][am][3];
        const float w0 = gwgt[tap][am][0], w1 = gwgt[tap][am][1];
        const float w2 = gwgt[tap][am][2], w3 = gwgt[tap][am][3];
        v4bf av;
#pragma unroll
        for (int u = 0; u < 4; ++u) {
            const float* xp = x + (size_t)xb + (size_t)(c0 + cl0 + u) * HW;
            const float v = w0 * xp[i0] + w1 * xp[i1] + w2 * xp[i2] + w3 * xp[i3];
            av[u] = (bf16_t)v;
        }
        *(v4bf*)&a_lds[buf][am][cl0] = av;              // one ds_store_b64
    };

    stageA(0, 0);

    for (int k = 0; k < KCHUNKS; ++k) {
        __syncthreads();                                 // chunk k staged
        if (k + 1 < KCHUNKS) stageA((k + 1) & 1, k + 1); // prefetch next chunk

        // A fragment: contiguous k = 16*half + s (matches B), 2x ds_load_b128
        const bf16_t* arow = &a_lds[k & 1][mw + l15][0];
        const v8bf alo = *(const v8bf*)(arow + (half << 4));
        const v8bf ahi = *(const v8bf*)(arow + (half << 4) + 8);
        const v16bf afrag = __builtin_shufflevector(
            alo, ahi, 0,1,2,3,4,5,6,7,8,9,10,11,12,13,14,15);

        // 4 n-subtiles: B fragment straight from L2 (16 contiguous bf16/lane)
        const size_t krow = ((size_t)k << 5) + (half << 4);
#pragma unroll
        for (int j = 0; j < 4; ++j) {
            const bf16_t* bp = Bp2 + (size_t)(nw0 + j * 16 + l15) * KDIM + krow;
            const v16bf bfrag = *(const v16bf*)bp;
            acc[j] = __builtin_amdgcn_wmma_f32_16x16x32_bf16(
                false, afrag, false, bfrag, (short)0, acc[j], false, false);
        }
    }

    // ---- epilogue: stage accumulators transposed, then coalesced b128 stores ----
    __syncthreads();
#pragma unroll
    for (int j = 0; j < 4; ++j) {
        const int n = nw0 + j * 16 + l15;
#pragma unroll
        for (int r = 0; r < 8; ++r)
            out_sT[n][mw + half * 8 + r] = acc[j][r];
    }
    __syncthreads();

#pragma unroll
    for (int u = 0; u < 8; ++u) {
        const int e  = tid + u * 256;        // 2048 vec4 stores total
        const int n  = e >> 3;
        const int m0 = (e & 7) << 2;
        const int s  = mbase + m0;
        const int b  = s / HW;
        const int rem = s - b * HW;
        const int ho = rem / Ww;
        const int wo = rem - ho * Ww;
        v4f v = *(const v4f*)&out_sT[n][m0];
        v = v + bias[n];
        *(v4f*)&out[((size_t)(b * Och + n) * Hh + ho) * Ww + wo] = v;
    }
}

// ---------------------------------------------------------------------------
extern "C" void kernel_launch(void* const* d_in, const int* in_sizes, int n_in,
                              void* d_out, int out_size, void* d_ws, size_t ws_size,
                              hipStream_t stream)
{
    const float* x        = (const float*)d_in[0];
    const float* offset_w = (const float*)d_in[1];
    const float* offset_b = (const float*)d_in[2];
    const float* deform_w = (const float*)d_in[3];
    const float* deform_b = (const float*)d_in[4];
    float* out = (float*)d_out;

    // workspace: offsets (f32) then packed bf16 weights
    float*  off = (float*)d_ws;                               // 2.65 MB
    bf16_t* Bp2 = (bf16_t*)((char*)d_ws +
                  (size_t)Bsz * OFF_CH * HW * sizeof(float)); // 1.18 MB

    offset_conv_kernel<<<dim3(Bsz * 12 * 3), 256, 0, stream>>>(x, offset_w, offset_b, off);
    pack_weights_kernel<<<(Och * KDIM + 255) / 256, 256, 0, stream>>>(deform_w, Bp2);
    deform_wmma_kernel<<<dim3(SPAT / 32), 256, 0, stream>>>(x, off, Bp2, deform_b, out);
}
